// SelfAttentionHead_30090540876176
// MI455X (gfx1250) — compile-verified, matrix operand order checked by reading, not run on hardware
//
#include <hip/hip_runtime.h>
#include <hip/hip_bf16.h>

// ---------------------------------------------------------------------------
// Self-attention head for MI455X (gfx1250, wave32, WMMA).
//   Q^T = Wq^T X^T, K^T = Wk^T X^T, V = X Wv   (f16 operands, f32 accum WMMA)
//   O = softmax(causal(Q K^T / sqrt(C))) V     (flash-attention, 16-row tiles)
// All fragments are built with 16-byte loads: for CDNA5 16-bit A/B layouts,
// lane -> row/col (lane%16) and elements map to two contiguous 8-half K runs
// at offsets 8*half and 16+8*half. Layouts below are chosen so the contraction
// dim is always contiguous in memory.
// ---------------------------------------------------------------------------

typedef __attribute__((ext_vector_type(16))) _Float16 v16h;
typedef __attribute__((ext_vector_type(8)))  _Float16 v8h;
typedef __attribute__((ext_vector_type(8)))  float    v8f;

#define B_    4
#define T_    4096
#define C_    1024
#define H_    64
#define NQT   (T_ / 16)         // query tiles per batch
#define SCALE 0.03125f          // C^-0.5 = 1024^-0.5 = 1/32

static __device__ __forceinline__ v8f vzero8() {
    v8f z = {0.f, 0.f, 0.f, 0.f, 0.f, 0.f, 0.f, 0.f};
    return z;
}

// Build a 16-half fragment from memory where the contraction dim is contiguous.
// p must already include the +8*hs lane-half offset; 16-byte aligned.
static __device__ __forceinline__ v16h ld_frag(const _Float16* p) {
    const v8h lo = *(const v8h*)(p);        // elements e=0..7  : k = 8hs+e
    const v8h hi = *(const v8h*)(p + 16);   // elements e=8..15 : k = 8hs+16+(e-8)
    return __builtin_shufflevector(lo, hi, 0, 1, 2, 3, 4, 5, 6, 7,
                                           8, 9, 10, 11, 12, 13, 14, 15);
}

// Same, from fp32 memory with on-the-fly f16 conversion (two 32B loads).
static __device__ __forceinline__ v16h ld_frag_f32(const float* p) {
    const v8f lo = *(const v8f*)(p);
    const v8f hi = *(const v8f*)(p + 16);
    v16h r;
#pragma unroll
    for (int e = 0; e < 8; ++e) {
        r[e]     = (_Float16)lo[e];
        r[e + 8] = (_Float16)hi[e];
    }
    return r;
}

// Pack 8 f32 accumulator rows into one 16-byte f16 store.
static __device__ __forceinline__ void st_rows(_Float16* p, v8f a) {
    v8h h;
#pragma unroll
    for (int r = 0; r < 8; ++r) h[r] = (_Float16)a[r];
    *(v8h*)p = h;
}

// reductions across the 16 lanes of one wave-half (xor masks 1,2,4,8 stay in half)
static __device__ __forceinline__ float half_max(float v) {
#pragma unroll
    for (int m = 1; m <= 8; m <<= 1) v = fmaxf(v, __shfl_xor(v, m, 32));
    return v;
}
static __device__ __forceinline__ float half_sum(float v) {
#pragma unroll
    for (int m = 1; m <= 8; m <<= 1) v += __shfl_xor(v, m, 32);
    return v;
}

static __device__ __forceinline__ v8f wmma_f16(v16h a, v16h b, v8f c) {
    return __builtin_amdgcn_wmma_f32_16x16x32_f16(false, a, false, b,
                                                  (short)0, c, false, false);
}

// ---------------------------------------------------------------------------
// Kernel 0: W fp32 [C][H] -> f16 transposed [3][H][C] (contraction contiguous).
// ---------------------------------------------------------------------------
__global__ void cvt_w_kernel(const float* __restrict__ Wq,
                             const float* __restrict__ Wk,
                             const float* __restrict__ Wv,
                             _Float16* __restrict__ W16t) {
    const int i = blockIdx.x * blockDim.x + threadIdx.x;   // i = h*C + c
    if (i < H_ * C_) {
        const int h = i / C_, c = i % C_;
        W16t[i]               = (_Float16)Wq[c * H_ + h];
        W16t[H_ * C_ + i]     = (_Float16)Wk[c * H_ + h];
        W16t[2 * H_ * C_ + i] = (_Float16)Wv[c * H_ + h];
    }
}

// ---------------------------------------------------------------------------
// Kernel 1: QKV projection, one wave per 16-row tile of X.
//   Q^T,K^T tiles: A = W^T frag, B = X frag  -> D(m=h, n=query)
//   V tiles:       A = X frag,  B = W^T frag -> D(m=query, n=h)
// X and W fragments have identical lane/element maps, so each is loaded once
// and reused in both roles. Q,K stored row-major [t][h]; V transposed [h][t].
// ---------------------------------------------------------------------------
__global__ void __launch_bounds__(128) qkv_kernel(const float* __restrict__ X,
                                                  const _Float16* __restrict__ W16t,
                                                  _Float16* __restrict__ QKV) {
    const int lane = threadIdx.x & 31;
    const int wave = (blockIdx.x * blockDim.x + threadIdx.x) >> 5;
    const int b    = wave / NQT;
    const int q0   = (wave % NQT) * 16;
    const int hs   = lane >> 4;
    const int ln   = lane & 15;

    _Float16* Q16  = QKV;                                 // [B][T][H]
    _Float16* K16  = QKV + (size_t)1 * B_ * T_ * H_;      // [B][T][H]
    _Float16* V16t = QKV + (size_t)2 * B_ * T_ * H_;      // [B][H][T]

    v8f accq[4], acck[4], accv[4];
#pragma unroll
    for (int t = 0; t < 4; ++t) { accq[t] = vzero8(); acck[t] = vzero8(); accv[t] = vzero8(); }

    const float* xrow = X + ((size_t)b * T_ + q0 + ln) * C_;

    for (int k0 = 0; k0 < C_; k0 += 32) {
        const v16h xf = ld_frag_f32(xrow + k0 + 8 * hs);  // X fragment (A and B role)
#pragma unroll
        for (int t = 0; t < 4; ++t) {
            const size_t wro = (size_t)(t * 16 + ln) * C_ + k0 + 8 * hs;
            const v16h wq = ld_frag(W16t + wro);
            const v16h wk = ld_frag(W16t + (size_t)H_ * C_ + wro);
            const v16h wv = ld_frag(W16t + (size_t)2 * H_ * C_ + wro);
            accq[t] = wmma_f16(wq, xf, accq[t]);   // Q^T tile
            acck[t] = wmma_f16(wk, xf, acck[t]);   // K^T tile
            accv[t] = wmma_f16(xf, wv, accv[t]);   // V tile
        }
    }

#pragma unroll
    for (int t = 0; t < 4; ++t) {
        // Q^T/K^T D: lane n = query (ln), rows r = h -> 8 contiguous h per lane
        _Float16* qp = Q16 + ((size_t)b * T_ + q0 + ln) * H_ + t * 16 + 8 * hs;
        _Float16* kp = K16 + ((size_t)b * T_ + q0 + ln) * H_ + t * 16 + 8 * hs;
        st_rows(qp, accq[t]);
        st_rows(kp, acck[t]);
        // V D: lane n = h, rows r = query -> 8 contiguous keys per lane
        _Float16* vp = V16t + ((size_t)b * H_ + t * 16 + ln) * T_ + q0 + 8 * hs;
        st_rows(vp, accv[t]);
    }
}

// ---------------------------------------------------------------------------
// Kernel 2: flash attention. One wave per 16-query tile, key blocks of 32.
// Online softmax; P transposed C/D->A layout via per-wave LDS (80B row stride,
// 16B aligned, conflict-free) + s_wait_dscnt for the same-wave RAW.
// ---------------------------------------------------------------------------
__global__ void __launch_bounds__(128) attn_kernel(const _Float16* __restrict__ QKV,
                                                   float* __restrict__ out) {
    const int lane  = threadIdx.x & 31;
    const int wslot = threadIdx.x >> 5;
    const int wave  = (blockIdx.x * blockDim.x + threadIdx.x) >> 5;
    const int b     = wave / NQT;
    const int qt    = wave % NQT;
    const int q0    = qt * 16;
    const int hs    = lane >> 4;
    const int ln    = lane & 15;

    const _Float16* Q16  = QKV;
    const _Float16* K16  = QKV + (size_t)1 * B_ * T_ * H_;
    const _Float16* V16t = QKV + (size_t)2 * B_ * T_ * H_;

    __shared__ __align__(16) _Float16 pbuf[4][16][40];   // 16x32 P tile, 80B stride

    // Q A-fragments (contraction = head dim, contiguous in row-major Q)
    const _Float16* qrow = Q16 + ((size_t)b * T_ + q0 + ln) * H_;
    const v16h aq0 = ld_frag(qrow + 8 * hs);
    const v16h aq1 = ld_frag(qrow + 32 + 8 * hs);

    float mrow[8], lrow[8];
    v8f   o[4];
#pragma unroll
    for (int r = 0; r < 8; ++r) { mrow[r] = -__builtin_inff(); lrow[r] = 0.f; }
#pragma unroll
    for (int t = 0; t < 4; ++t) o[t] = vzero8();

    const int nkb = (q0 + 15) / 32 + 1;       // causal: key blocks up to diagonal
    for (int kb = 0; kb < nkb; ++kb) {
        const int kbase = kb * 32;

        // S tiles for keys [kbase..+15] and [kbase+16..+31]
        v8f s0 = vzero8(), s1 = vzero8();
        const _Float16* krow0 = K16 + ((size_t)b * T_ + kbase + ln) * H_;
        const _Float16* krow1 = krow0 + (size_t)16 * H_;
#pragma unroll
        for (int hstep = 0; hstep < 2; ++hstep) {
            const v16h kf0 = ld_frag(krow0 + hstep * 32 + 8 * hs);
            const v16h kf1 = ld_frag(krow1 + hstep * 32 + 8 * hs);
            const v16h aq  = hstep ? aq1 : aq0;
            s0 = wmma_f16(aq, kf0, s0);
            s1 = wmma_f16(aq, kf1, s1);
        }

        // scale + causal mask + online softmax (row stats live per wave-half)
#pragma unroll
        for (int r = 0; r < 8; ++r) {
            const int qi = q0 + r + hs * 8;
            float v0 = s0[r] * SCALE;
            float v1 = s1[r] * SCALE;
            if (kbase + ln > qi)      v0 = -__builtin_inff();
            if (kbase + 16 + ln > qi) v1 = -__builtin_inff();

            const float rm   = half_max(fmaxf(v0, v1));
            const float newm = fmaxf(mrow[r], rm);
            const float cf   = __expf(mrow[r] - newm);
            const float p0   = __expf(v0 - newm);
            const float p1   = __expf(v1 - newm);
            const float rs   = half_sum(p0 + p1);
            mrow[r] = newm;
            lrow[r] = lrow[r] * cf + rs;
#pragma unroll
            for (int t = 0; t < 4; ++t) o[t][r] *= cf;

            pbuf[wslot][r + hs * 8][ln]      = (_Float16)p0;
            pbuf[wslot][r + hs * 8][16 + ln] = (_Float16)p1;
        }

        // same-wave LDS RAW: drain DS counter before transposed re-read
        __asm volatile("s_wait_dscnt 0x0" ::: "memory");

        // P as A-fragment: two ds_load_b128 per lane
        const v16h pf = ld_frag(&pbuf[wslot][ln][8 * hs]);

        // O += P @ V (contraction = 32 keys, contiguous in transposed V)
#pragma unroll
        for (int t = 0; t < 4; ++t) {
            const v16h vf =
                ld_frag(V16t + ((size_t)b * H_ + t * 16 + ln) * T_ + kbase + 8 * hs);
            o[t] = wmma_f16(pf, vf, o[t]);
        }
    }

    // normalize and write fp32 output
#pragma unroll
    for (int r = 0; r < 8; ++r) {
        const float inv = 1.0f / lrow[r];
#pragma unroll
        for (int t = 0; t < 4; ++t)
            out[((size_t)b * T_ + q0 + r + hs * 8) * H_ + t * 16 + ln] =
                o[t][r] * inv;
    }
}

// ---------------------------------------------------------------------------
extern "C" void kernel_launch(void* const* d_in, const int* in_sizes, int n_in,
                              void* d_out, int out_size, void* d_ws, size_t ws_size,
                              hipStream_t stream) {
    const float* X  = (const float*)d_in[0];
    const float* Wq = (const float*)d_in[1];
    const float* Wk = (const float*)d_in[2];
    const float* Wv = (const float*)d_in[3];
    float* out = (float*)d_out;

    // workspace: [3][H][C] f16 W^T (384 KB) then Q[B][T][H], K[B][T][H],
    // V^T[B][H][T] f16 (6 MB)
    _Float16* W16t = (_Float16*)d_ws;
    _Float16* QKV  = W16t + (size_t)3 * H_ * C_;

    cvt_w_kernel<<<(H_ * C_ + 255) / 256, 256, 0, stream>>>(Wq, Wk, Wv, W16t);

    const int nwaves = B_ * NQT;               // 1024 waves
    const int blocks = nwaves * 32 / 128;      // 4 waves / block
    qkv_kernel<<<blocks, 128, 0, stream>>>(X, W16t, QKV);
    attn_kernel<<<blocks, 128, 0, stream>>>(QKV, out);
}
